// GCN_5772436046127
// MI455X (gfx1250) — matrix operand order repaired and checked by reading, not compiled, not run on hardware
//
#include <hip/hip_runtime.h>
#include <hip/hip_bf16.h>
#include <stdint.h>

// Problem constants (match reference)
#define NN      100000
#define EE      1000000
#define D_IN    128
#define D_H     64
#define D_OUT   64
#define MTILES  (NN / 16)      // 6250, exact

typedef __attribute__((ext_vector_type(16))) __bf16 v16bf;
typedef __attribute__((ext_vector_type(8)))  float  v8f;
typedef __attribute__((ext_vector_type(4)))  float  v4f;

// ---------------- degree / normalization ----------------
__global__ void k_deg_init(float* __restrict__ deg, int n) {
    int i = blockIdx.x * blockDim.x + threadIdx.x;
    if (i < n) deg[i] = 1.0f;                    // self-loop contributes 1
}

__global__ void k_deg_count(const int* __restrict__ dst, float* __restrict__ deg, int e) {
    int i = blockIdx.x * blockDim.x + threadIdx.x;
    if (i < e) atomicAdd(&deg[dst[i]], 1.0f);
}

__global__ void k_dinv(float* __restrict__ deg, int n) {
    int i = blockIdx.x * blockDim.x + threadIdx.x;
    if (i < n) deg[i] = rsqrtf(deg[i]);          // deg >= 1 always (self-loops)
}

// ---------------- W pre-swizzle into WMMA B-fragment order ----------------
// Output element o = ((q*4 + t)*32 + lane)*16 + i  holds
//   W[k = q*32 + (i>=8 ? 16 : 0) + (lane>=16 ? 8 : 0) + (i&7)][c = t*16 + (lane&15)]  as bf16,
// so each lane's B fragment (v16bf) is one contiguous 32B load.
__global__ void k_swizzle_w(const float* __restrict__ W, __bf16* __restrict__ Wb, int K) {
    int o = blockIdx.x * blockDim.x + threadIdx.x;
    if (o >= K * 64) return;
    int i = o & 15;
    int l = (o >> 4) & 31;
    int t = (o >> 9) & 3;
    int q = o >> 11;
    int k = q * 32 + ((i >> 3) << 4) + ((l >> 4) << 3) + (i & 7);
    int c = t * 16 + (l & 15);
    Wb[o] = (__bf16)W[k * 64 + c];
}

// ---------------- WMMA GEMM: C[N,64] = op(A[N,K]) @ W[K,64] ----------------
// One wave per 16-row M tile; 4 accumulators cover all 64 output columns.
template<bool RELU>
__global__ __launch_bounds__(256)
void k_gemm_wmma(const float* __restrict__ A, const __bf16* __restrict__ Wb,
                 float* __restrict__ C, int K, int mtiles) {
    int wave = blockIdx.x * (blockDim.x >> 5) + (threadIdx.x >> 5);
    if (wave >= mtiles) return;                  // wave-uniform guard (EXEC stays all-1s)
    int lane = threadIdx.x & 31;
    int half = lane >> 4;                        // 0: K 0..7/16..23   1: K 8..15/24..31
    int mrow = wave * 16 + (lane & 15);
    const float* arow = A + (size_t)mrow * (size_t)K;
    const v16bf* bv = (const v16bf*)Wb;

    v8f acc0 = {}, acc1 = {}, acc2 = {}, acc3 = {};

    for (int kk = 0; kk < K; kk += 32) {
        const float* p = arow + kk + half * 8;
        v4f f0 = *(const v4f*)(p);
        v4f f1 = *(const v4f*)(p + 4);
        v4f f2 = *(const v4f*)(p + 16);
        v4f f3 = *(const v4f*)(p + 20);

        v16bf a;
#pragma unroll
        for (int i = 0; i < 4; ++i) {
            float x0 = f0[i], x1 = f1[i], x2 = f2[i], x3 = f3[i];
            if (RELU) { x0 = fmaxf(x0, 0.f); x1 = fmaxf(x1, 0.f);
                        x2 = fmaxf(x2, 0.f); x3 = fmaxf(x3, 0.f); }
            a[i]      = (__bf16)x0;   // K = kk + half*8 + i
            a[4 + i]  = (__bf16)x1;   // K = kk + half*8 + 4 + i
            a[8 + i]  = (__bf16)x2;   // K = kk + 16 + half*8 + i
            a[12 + i] = (__bf16)x3;   // K = kk + 16 + half*8 + 4 + i
        }

        int q = kk >> 5;
        v16bf b0 = bv[(q * 4 + 0) * 32 + lane];
        v16bf b1 = bv[(q * 4 + 1) * 32 + lane];
        v16bf b2 = bv[(q * 4 + 2) * 32 + lane];
        v16bf b3 = bv[(q * 4 + 3) * 32 + lane];

        acc0 = __builtin_amdgcn_wmma_f32_16x16x32_bf16(false, a, false, b0, (short)0, acc0, false, false);
        acc1 = __builtin_amdgcn_wmma_f32_16x16x32_bf16(false, a, false, b1, (short)0, acc1, false, false);
        acc2 = __builtin_amdgcn_wmma_f32_16x16x32_bf16(false, a, false, b2, (short)0, acc2, false, false);
        acc3 = __builtin_amdgcn_wmma_f32_16x16x32_bf16(false, a, false, b3, (short)0, acc3, false, false);
    }

    // C/D layout: VGPR j -> row m0 + half*8 + j, col = lane&15 within each 16-col tile
    int col = lane & 15;
    size_t rbase = (size_t)(wave * 16 + half * 8);
#pragma unroll
    for (int j = 0; j < 8; ++j) {
        float* crow = C + (rbase + j) * 64;
        crow[col]      = acc0[j];
        crow[16 + col] = acc1[j];
        crow[32 + col] = acc2[j];
        crow[48 + col] = acc3[j];
    }
}

// ---------------- aggregation ----------------
// out[n][f] = bias[f] + H[n][f] * dinv[n]^2   (self-loop term, also inits for atomics)
__global__ void k_agg_init(const float* __restrict__ H, const float* __restrict__ dinv,
                           const float* __restrict__ bias, float* __restrict__ out, int n) {
    int idx = blockIdx.x * blockDim.x + threadIdx.x;
    if (idx >= n * 64) return;
    int node = idx >> 6;
    int f    = idx & 63;
    float di = dinv[node];
    out[idx] = bias[f] + H[idx] * di * di;
}

// one lane per (edge, feature-pair): lanes 0..31 of each edge handle f and f+32
__global__ void k_agg_edges(const float* __restrict__ H, const int* __restrict__ src,
                            const int* __restrict__ dst, const float* __restrict__ dinv,
                            float* __restrict__ out, int e) {
    int idx  = blockIdx.x * blockDim.x + threadIdx.x;
    int edge = idx >> 5;
    if (edge >= e) return;
    int f = idx & 31;
    int s = src[edge];
    int d = dst[edge];
    float nrm = dinv[s] * dinv[d];
    const float* hs = H + (size_t)s * 64;
    float* od = out + (size_t)d * 64;
    atomicAdd(&od[f],      hs[f]      * nrm);
    atomicAdd(&od[f + 32], hs[f + 32] * nrm);
}

// ---------------- launcher ----------------
static inline size_t align256(size_t x) { return (x + 255) & ~(size_t)255; }

extern "C" void kernel_launch(void* const* d_in, const int* in_sizes, int n_in,
                              void* d_out, int out_size, void* d_ws, size_t ws_size,
                              hipStream_t stream) {
    (void)in_sizes; (void)n_in; (void)out_size; (void)ws_size;

    const float* x     = (const float*)d_in[0];        // [N,128]
    const int*   edges = (const int*)  d_in[1];        // [2,E]
    const float* W1    = (const float*)d_in[2];        // [128,64]
    const float* b1    = (const float*)d_in[3];        // [64]
    const float* W2    = (const float*)d_in[4];        // [64,64]
    const float* b2    = (const float*)d_in[5];        // [64]
    float*       out   = (float*)d_out;                // [N,64]

    const int* src = edges;          // edge_index[0]
    const int* dst = edges + EE;     // edge_index[1]

    // workspace carve-up
    char* ws = (char*)d_ws;
    size_t off = 0;
    float*  dinv = (float*)(ws + off);  off += align256((size_t)NN * 4);
    float*  bufA = (float*)(ws + off);  off += align256((size_t)NN * 64 * 4);
    float*  bufB = (float*)(ws + off);  off += align256((size_t)NN * 64 * 4);
    __bf16* Wb1  = (__bf16*)(ws + off); off += align256((size_t)D_IN * 64 * 2);
    __bf16* Wb2  = (__bf16*)(ws + off); off += align256((size_t)D_H  * 64 * 2);

    const int T = 256;

    // 1) normalization coefficients
    k_deg_init <<<(NN + T - 1) / T, T, 0, stream>>>(dinv, NN);
    k_deg_count<<<(EE + T - 1) / T, T, 0, stream>>>(dst, dinv, EE);
    k_dinv     <<<(NN + T - 1) / T, T, 0, stream>>>(dinv, NN);

    // 2) weight pre-swizzle to WMMA B-fragment order (bf16)
    k_swizzle_w<<<(D_IN * 64 + T - 1) / T, T, 0, stream>>>(W1, Wb1, D_IN);
    k_swizzle_w<<<(D_H  * 64 + T - 1) / T, T, 0, stream>>>(W2, Wb2, D_H);

    int gemm_blocks = (MTILES + 7) / 8;   // 8 waves/block

    // 3) layer 1: H1 = X @ W1
    k_gemm_wmma<false><<<gemm_blocks, T, 0, stream>>>(x, Wb1, bufA, D_IN, MTILES);
    // 4) aggregate: bufB = b1 + A_hat-weighted sum of bufA
    k_agg_init <<<((NN * 64) + T - 1) / T, T, 0, stream>>>(bufA, dinv, b1, bufB, NN);
    k_agg_edges<<<((EE * 32) + T - 1) / T, T, 0, stream>>>(bufA, src, dst, dinv, bufB, EE);

    // 5) layer 2: H2 = relu(bufB) @ W2   (ReLU fused into A-fragment load)
    k_gemm_wmma<true><<<gemm_blocks, T, 0, stream>>>(bufB, Wb2, bufA, D_H, MTILES);
    // 6) aggregate into final output
    k_agg_init <<<((NN * 64) + T - 1) / T, T, 0, stream>>>(bufA, dinv, b2, out, NN);
    k_agg_edges<<<((EE * 32) + T - 1) / T, T, 0, stream>>>(bufA, src, dst, dinv, out, EE);
}
